// RenderNet_40638980555479
// MI455X (gfx1250) — matrix operand order repaired and checked by reading, not compiled, not run on hardware
//
#include <hip/hip_runtime.h>
#include <math.h>

typedef __attribute__((ext_vector_type(16))) _Float16 v16h;
typedef __attribute__((ext_vector_type(8)))  _Float16 v8h;
typedef __attribute__((ext_vector_type(8)))  float    v8f;

#define DD 32
#define HH 256
#define WW 384
#define HW (HH*WW)            // 98,304
#define VOX (DD*HH*WW)        // 3,145,728
#define NV 2
#define XBLK 128              // pixels per block (8 waves x 16)
#define XS 130                // staged row length (XBLK + 2 halo)

// ---- workspace layout (bytes) ----
#define OFF_HPSV   0                         // 64*9 f32
#define OFF_HFWD   4096                      // 32*9 f32
#define OFF_DISP   8192                      // 32 f32
#define OFF_WP1    12288                     // 7 chunks * 512 halfs
#define OFF_WP2    20480
#define OFF_WP3    28672
#define OFF_PSV    65536                     // f16 [8][D][H][W] (ch 6,7 zero)
#define OFF_H1     (OFF_PSV + 8u*VOX*2u)     // f16 [8][D][H][W]
#define OFF_H2     (OFF_H1 + 8u*VOX*2u)      // f16 [8][D][H][W]
#define OFF_PRED   (OFF_H2 + 8u*VOX*2u)      // f32 [3][D][H][W]
#define OFF_RGBA   OFF_H2                    // f32 [D][H][W][4] (aliases h2, same size)
#define OFF_WARP   OFF_H1                    // f32 [D][H][W][4] (aliases h1)

// ----------------------------------------------------------------------------
// 3x3 helpers
// ----------------------------------------------------------------------------
__device__ inline void mat3_mul(const float* a, const float* b, float* c) {
  for (int i = 0; i < 3; ++i)
    for (int j = 0; j < 3; ++j)
      c[i*3+j] = a[i*3+0]*b[0*3+j] + a[i*3+1]*b[1*3+j] + a[i*3+2]*b[2*3+j];
}

__device__ inline void mat3_inv(const float* m, float* o) {
  float c00 =   m[4]*m[8] - m[5]*m[7];
  float c01 = -(m[3]*m[8] - m[5]*m[6]);
  float c02 =   m[3]*m[7] - m[4]*m[6];
  float det = m[0]*c00 + m[1]*c01 + m[2]*c02;
  float id  = 1.0f / det;
  o[0] =  c00*id;
  o[1] = -(m[1]*m[8] - m[2]*m[7])*id;
  o[2] =  (m[1]*m[5] - m[2]*m[4])*id;
  o[3] =  c01*id;
  o[4] =  (m[0]*m[8] - m[2]*m[6])*id;
  o[5] = -(m[0]*m[5] - m[2]*m[3])*id;
  o[6] =  c02*id;
  o[7] = -(m[0]*m[7] - m[1]*m[6])*id;
  o[8] =  (m[0]*m[4] - m[1]*m[3])*id;
}

// ----------------------------------------------------------------------------
// Kernel 0: geometry setup (single thread)
// ----------------------------------------------------------------------------
__global__ void setup_kernel(const float* __restrict__ src_exts,
                             const float* __restrict__ src_ints,
                             const float* __restrict__ tgt_ext,
                             const float* __restrict__ tgt_int,
                             float* __restrict__ Hpsv,
                             float* __restrict__ Hfwd,
                             float* __restrict__ dispTab)
{
  float Rinv[9], tinv[3];
  for (int i = 0; i < 3; ++i)
    for (int j = 0; j < 3; ++j)
      Rinv[i*3+j] = src_exts[j*4+i];
  for (int i = 0; i < 3; ++i)
    tinv[i] = -(Rinv[i*3+0]*src_exts[0*4+3] + Rinv[i*3+1]*src_exts[1*4+3] +
                Rinv[i*3+2]*src_exts[2*4+3]);

  float Kinv[9];
  mat3_inv(src_ints, Kinv);

  for (int n = 0; n < NV; ++n) {
    const float* E = src_exts + n*16;
    float R[9], t[3];
    for (int i = 0; i < 3; ++i) {
      for (int j = 0; j < 3; ++j)
        R[i*3+j] = E[i*4+0]*Rinv[0*3+j] + E[i*4+1]*Rinv[1*3+j] + E[i*4+2]*Rinv[2*3+j];
      t[i] = E[i*4+0]*tinv[0] + E[i*4+1]*tinv[1] + E[i*4+2]*tinv[2] + E[i*4+3];
    }
    const float* Ks = src_ints + n*9;
    for (int d = 0; d < DD; ++d) {
      float invd = 0.01f + (0.99f/31.0f)*(float)d;
      float M[9];
      for (int k = 0; k < 9; ++k) M[k] = R[k];
      M[2] += t[0]*invd; M[5] += t[1]*invd; M[8] += t[2]*invd;
      float T1[9], Hm[9];
      mat3_mul(Ks, M, T1);
      mat3_mul(T1, Kinv, Hm);
      for (int k = 0; k < 9; ++k) Hpsv[(d*NV+n)*9+k] = Hm[k];
    }
  }
  {
    float R[9], t[3];
    for (int i = 0; i < 3; ++i) {
      for (int j = 0; j < 3; ++j)
        R[i*3+j] = tgt_ext[i*4+0]*Rinv[0*3+j] + tgt_ext[i*4+1]*Rinv[1*3+j] +
                   tgt_ext[i*4+2]*Rinv[2*3+j];
      t[i] = tgt_ext[i*4+0]*tinv[0] + tgt_ext[i*4+1]*tinv[1] +
             tgt_ext[i*4+2]*tinv[2] + tgt_ext[i*4+3];
    }
    for (int d = 0; d < DD; ++d) {
      float invd = 0.01f + (0.99f/31.0f)*(float)d;
      float M[9];
      for (int k = 0; k < 9; ++k) M[k] = R[k];
      M[2] += t[0]*invd; M[5] += t[1]*invd; M[8] += t[2]*invd;
      float T1[9], Hm[9];
      mat3_mul(tgt_int, M, T1);
      mat3_mul(T1, Kinv, Hm);
      for (int k = 0; k < 9; ++k) Hfwd[d*9+k] = Hm[k];
      dispTab[d] = invd;
    }
  }
}

// ----------------------------------------------------------------------------
// Kernel 1: pack weights into WMMA A-fragment layout (f16), K-slot = pos*8+ic.
// A 16x32 f16 layout (ISA 7.12.2): lane row M=lane%16; element j ->
//   K = j + (j>=8 ? 8 : 0) + (lane>=16 ? 8 : 0) within the chunk.
// K-slot decode: ic = k&7, pos = k>>3 (pos = kd*9+kh*3+kw, valid pos<27).
// ----------------------------------------------------------------------------
__global__ void pack_weights_kernel(const float* __restrict__ w,  // [OC][IC][27]
                                    _Float16* __restrict__ packed,
                                    int OC, int IC)
{
  int t = blockIdx.x*blockDim.x + threadIdx.x;
  if (t >= 7*32) return;
  int ch = t >> 5, lane = t & 31;
  int lm = lane & 15;
  int hs = (lane >> 4) * 8;
  for (int j = 0; j < 16; ++j) {
    int k = ch*32 + ((j < 8) ? (j + hs) : (j + 8 + hs));
    int ic  = k & 7;
    int pos = k >> 3;
    float v = 0.0f;
    if (lm < OC && ic < IC && pos < 27) v = w[(lm*IC + ic)*27 + pos];
    packed[(ch*32 + lane)*16 + j] = (_Float16)v;
  }
}

// ----------------------------------------------------------------------------
// Kernel 2: plane-sweep volume via inverse warp + bilinear gather (f16).
// ----------------------------------------------------------------------------
__global__ void psv_warp_kernel(const float* __restrict__ src,   // [2][3][H][W]
                                const float* __restrict__ Hpsv,
                                _Float16* __restrict__ psv)      // [8][D][H][W]
{
  int idx = blockIdx.x*blockDim.x + threadIdx.x;
  if (idx >= DD*NV*HW) return;
  int p  = idx % HW;
  int dn = idx / HW;
  int n  = dn % NV;
  float fx = (float)(p % WW);
  float fy = (float)(p / WW);
  const float* Hm = Hpsv + dn*9;
  float px = Hm[0]*fx + Hm[1]*fy + Hm[2];
  float py = Hm[3]*fx + Hm[4]*fy + Hm[5];
  float pz = Hm[6]*fx + Hm[7]*fy + Hm[8];
  if (pz == 0.0f) pz = 1e-8f;
  float sx = px/pz, sy = py/pz;
  float x0f = floorf(sx), y0f = floorf(sy);
  int   x0  = (int)x0f,  y0  = (int)y0f;
  float ax = sx - x0f, ay = sy - y0f;

  float wt[4] = { (1.0f-ax)*(1.0f-ay), (1.0f-ax)*ay, ax*(1.0f-ay), ax*ay };
  int   xs[4] = { x0, x0, x0+1, x0+1 };
  int   ys[4] = { y0, y0+1, y0, y0+1 };

  const float* img = src + n*3*HW;
  float a0 = 0.0f, a1 = 0.0f, a2 = 0.0f;
  for (int c4 = 0; c4 < 4; ++c4) {
    if ((unsigned)xs[c4] < (unsigned)WW && (unsigned)ys[c4] < (unsigned)HH) {
      int o = ys[c4]*WW + xs[c4];
      float w = wt[c4];
      a0 += (img[o]        + 1.0f) * w;
      a1 += (img[HW + o]   + 1.0f) * w;
      a2 += (img[2*HW + o] + 1.0f) * w;
    }
  }
  int dnbase = (dn / NV) * HW + p;
  psv[(n*3+0)*VOX + dnbase] = (_Float16)(a0 - 1.0f);
  psv[(n*3+1)*VOX + dnbase] = (_Float16)(a1 - 1.0f);
  psv[(n*3+2)*VOX + dnbase] = (_Float16)(a2 - 1.0f);
}

__global__ void zero_half_kernel(_Float16* __restrict__ p, int n)
{
  int i = blockIdx.x*blockDim.x + threadIdx.x;
  if (i < n) p[i] = (_Float16)0.0f;
}

__global__ void zero_kernel(float* __restrict__ p, int n)
{
  int i = blockIdx.x*blockDim.x + threadIdx.x;
  if (i < n) p[i] = 0.0f;
}

// ----------------------------------------------------------------------------
// Kernel 3: 3x3x3 conv, IC fixed at 8 (zero-padded), as LDS-staged WMMA GEMM.
// Block = 256 threads = 8 waves; block covers 128 pixels along W at (tz,ty).
// LDS slab: smem[zy 0..9][x 0..129][ic 0..7], row zy=9 is the zero pad row
// (K-padding slot pos=27 maps there).  Each lane's B half-fragment (8 f16,
// consecutive K-slots = one pos, ic 0..7) is a single 16-byte LDS load.
// ----------------------------------------------------------------------------
template<int OC, int ACT>
__global__ void __launch_bounds__(256)
conv3d_wmma_kernel(const _Float16* __restrict__ in,     // [8][D][H][W]
                   const _Float16* __restrict__ wpack,  // [7][32][16]
                   const float*    __restrict__ bias,   // [OC]
                   void* __restrict__ outv)             // [OC][D][H][W]
{
  __shared__ __align__(16) _Float16 smem[10*XS*8];

  int bx = blockIdx.x % (WW/XBLK);
  int rr = blockIdx.x / (WW/XBLK);
  int ty = rr % HH;
  int tz = rr / HH;
  int xblk = bx * XBLK;
  int tid = (int)threadIdx.x;

  // stage rows r = ic*9 + zy (coalesced along x), halo/oob -> 0
  for (int i = tid; i < 72*XS; i += 256) {
    int r  = i / XS;
    int x  = i - r*XS;
    int ic = r / 9;
    int zy = r - ic*9;
    int dz = zy / 3;
    int dy = zy - dz*3;
    int zz = tz + dz - 1;
    int yy = ty + dy - 1;
    int xx = xblk + x - 1;
    _Float16 v = (_Float16)0.0f;
    if ((unsigned)zz < (unsigned)DD && (unsigned)yy < (unsigned)HH &&
        (unsigned)xx < (unsigned)WW)
      v = in[((ic*DD + zz)*HH + yy)*WW + xx];
    smem[(zy*XS + x)*8 + ic] = v;
  }
  // zero pad row (zy == 9)
  for (int i = tid; i < XS*8; i += 256)
    smem[9*XS*8 + i] = (_Float16)0.0f;
  __syncthreads();

  int lane   = tid & 31;
  int waveId = tid >> 5;
  int lm     = lane & 15;
  int upper  = lane >> 4;
  int lx     = waveId*16 + lm;   // local pixel (0..127); LDS x = lx + kw

  v8f acc = {};
#pragma unroll
  for (int ch = 0; ch < 7; ++ch) {
    v16h a = *(const v16h*)(wpack + (ch*32 + lane)*16);
    // B element j -> K = ch*32 + j + (j>=8 ? 8 : 0) + upper*8; pos = K>>3; first 8 = pos0,
    // second 8 = pos0+2.  Row offsets constant-fold; 'upper' picks one.
    const int pA0 = ch*4 + 0, pA1 = ch*4 + 2;   // upper == 0
    const int pB0 = ch*4 + 1, pB1 = ch*4 + 3;   // upper == 1
    int o0 = upper ? ((pB0/3)*XS + (pB0%3)) : ((pA0/3)*XS + (pA0%3));
    int o1 = upper ? ((pB1/3)*XS + (pB1%3)) : ((pA1/3)*XS + (pA1%3));
    v8h blo = *(const v8h*)(smem + (o0 + lx)*8);
    v8h bhi = *(const v8h*)(smem + (o1 + lx)*8);
    v16h b = __builtin_shufflevector(blo, bhi,
                                     0,1,2,3,4,5,6,7,8,9,10,11,12,13,14,15);
    acc = __builtin_amdgcn_wmma_f32_16x16x32_f16(false, a, false, b,
                                                 (short)0, acc, false, false);
  }

  // C/D layout: VGPR r -> M = r + 8*upper, N = lane%16
  int mbase = upper*8;
  int px = xblk + lx;
#pragma unroll
  for (int r = 0; r < 8; ++r) {
    int oc = r + mbase;
    if (oc < OC) {
      float v = acc[r] + bias[oc];
      int o = ((oc*DD + tz)*HH + ty)*WW + px;
      if (ACT == 0) {
        ((_Float16*)outv)[o] = (_Float16)fmaxf(v, 0.0f);
      } else {
        ((float*)outv)[o] = 1.0f / (1.0f + __expf(-v));
      }
    }
  }
}

// ----------------------------------------------------------------------------
// Kernel 4: combine -> rgba [D][H][W][4]
// ----------------------------------------------------------------------------
__global__ void combine_kernel(const float* __restrict__ pred,
                               const _Float16* __restrict__ psv,
                               float* __restrict__ rgba)
{
  int idx = blockIdx.x*blockDim.x + threadIdx.x;   // d*HW + p
  if (idx >= VOX) return;
  float alpha = pred[idx];
  float w0 = pred[VOX   + idx];
  float w1 = pred[2*VOX + idx];
  float r = w0*(float)psv[idx]          + w1*(float)psv[3*VOX + idx];
  float g = w0*(float)psv[VOX + idx]    + w1*(float)psv[4*VOX + idx];
  float b = w0*(float)psv[2*VOX + idx]  + w1*(float)psv[5*VOX + idx];
  float* o = rgba + idx*4;
  o[0] = r; o[1] = g; o[2] = b; o[3] = alpha;
}

// ----------------------------------------------------------------------------
// Kernel 5: forward projective splat (scatter-add, 4 bilinear corners)
// ----------------------------------------------------------------------------
__global__ void splat_kernel(const float* __restrict__ rgba,
                             const float* __restrict__ Hfwd,
                             float* __restrict__ warp)
{
  int idx = blockIdx.x*blockDim.x + threadIdx.x;   // d*HW + p
  if (idx >= VOX) return;
  int p = idx % HW;
  int d = idx / HW;
  float fx = (float)(p % WW);
  float fy = (float)(p / WW);
  const float* Hm = Hfwd + d*9;
  float px = Hm[0]*fx + Hm[1]*fy + Hm[2];
  float py = Hm[3]*fx + Hm[4]*fy + Hm[5];
  float pz = Hm[6]*fx + Hm[7]*fy + Hm[8];
  if (pz == 0.0f) pz = 1e-8f;
  float sx = px/pz, sy = py/pz;
  float x0f = floorf(sx), y0f = floorf(sy);
  int   x0  = (int)x0f,  y0  = (int)y0f;
  float ax = sx - x0f, ay = sy - y0f;

  float vr = rgba[idx*4+0], vg = rgba[idx*4+1], vb = rgba[idx*4+2], va = rgba[idx*4+3];

  float wt[4] = { (1.0f-ax)*(1.0f-ay), (1.0f-ax)*ay, ax*(1.0f-ay), ax*ay };
  int   xs[4] = { x0, x0, x0+1, x0+1 };
  int   ys[4] = { y0, y0+1, y0, y0+1 };
  for (int c4 = 0; c4 < 4; ++c4) {
    if ((unsigned)xs[c4] < (unsigned)WW && (unsigned)ys[c4] < (unsigned)HH) {
      float w = wt[c4];
      int o = (d*HW + ys[c4]*WW + xs[c4]) * 4;
      atomicAdd(&warp[o+0], vr*w);
      atomicAdd(&warp[o+1], vg*w);
      atomicAdd(&warp[o+2], vb*w);
      atomicAdd(&warp[o+3], va*w);
    }
  }
}

// ----------------------------------------------------------------------------
// Kernel 6: back-to-front over-composite; writes final outputs
// out layout: [3*HW rendering (CHW)] [HW acc_alpha] [HW disp]
// ----------------------------------------------------------------------------
__global__ void composite_kernel(const float* __restrict__ warp,
                                 const float* __restrict__ dispTab,
                                 float* __restrict__ out)
{
  int p = blockIdx.x*blockDim.x + threadIdx.x;
  if (p >= HW) return;
  float T = 1.0f, r = 0.0f, g = 0.0f, b = 0.0f, acc = 0.0f, dsp = 0.0f;
  for (int d = DD-1; d >= 0; --d) {
    const float* q = warp + (d*HW + p)*4;
    float a  = q[3];
    float wt = a * T;
    r   += wt * q[0];
    g   += wt * q[1];
    b   += wt * q[2];
    acc += wt;
    dsp += wt * dispTab[d];
    T *= (1.0f - a);
  }
  out[p]        = r;
  out[HW + p]   = g;
  out[2*HW + p] = b;
  out[3*HW + p] = acc;
  out[4*HW + p] = dsp;
}

// ----------------------------------------------------------------------------
// launch
// ----------------------------------------------------------------------------
extern "C" void kernel_launch(void* const* d_in, const int* in_sizes, int n_in,
                              void* d_out, int out_size, void* d_ws, size_t ws_size,
                              hipStream_t stream)
{
  const float* src_imgs = (const float*)d_in[0];
  const float* src_exts = (const float*)d_in[1];
  const float* src_ints = (const float*)d_in[2];
  const float* tgt_ext  = (const float*)d_in[3];
  const float* tgt_int  = (const float*)d_in[4];
  const float* w1 = (const float*)d_in[5];
  const float* b1 = (const float*)d_in[6];
  const float* w2 = (const float*)d_in[7];
  const float* b2 = (const float*)d_in[8];
  const float* w3 = (const float*)d_in[9];
  const float* b3 = (const float*)d_in[10];

  char* ws = (char*)d_ws;
  float*    Hpsv    = (float*)(ws + OFF_HPSV);
  float*    Hfwd    = (float*)(ws + OFF_HFWD);
  float*    dispTab = (float*)(ws + OFF_DISP);
  _Float16* wp1     = (_Float16*)(ws + OFF_WP1);
  _Float16* wp2     = (_Float16*)(ws + OFF_WP2);
  _Float16* wp3     = (_Float16*)(ws + OFF_WP3);
  _Float16* psvbuf  = (_Float16*)(ws + OFF_PSV);
  _Float16* h1buf   = (_Float16*)(ws + OFF_H1);
  _Float16* h2buf   = (_Float16*)(ws + OFF_H2);
  float*    predbuf = (float*)(ws + OFF_PRED);
  float*    rgbabuf = (float*)(ws + OFF_RGBA);
  float*    warpbuf = (float*)(ws + OFF_WARP);

  // 0) geometry
  setup_kernel<<<1, 1, 0, stream>>>(src_exts, src_ints, tgt_ext, tgt_int,
                                    Hpsv, Hfwd, dispTab);

  // 1) weight packing (all layers use 7 K-chunks, K-slot = pos*8 + ic)
  pack_weights_kernel<<<1, 256, 0, stream>>>(w1, wp1, 8, 6);
  pack_weights_kernel<<<1, 256, 0, stream>>>(w2, wp2, 8, 8);
  pack_weights_kernel<<<1, 256, 0, stream>>>(w3, wp3, 3, 8);

  // 2) plane-sweep volume (f16, channels 6..7 zeroed)
  {
    int n = DD*NV*HW;
    psv_warp_kernel<<<(n + 255)/256, 256, 0, stream>>>(src_imgs, Hpsv, psvbuf);
    int nz = 2*VOX;
    zero_half_kernel<<<(nz + 255)/256, 256, 0, stream>>>(psvbuf + 6*VOX, nz);
  }

  // 3) conv net: 6->8 relu, 8->8 relu, 8->3 sigmoid (LDS-staged WMMA)
  {
    int blocks = DD*HH*(WW/XBLK);   // 24,576 blocks of 256 threads (8 waves)
    conv3d_wmma_kernel<8, 0><<<blocks, 256, 0, stream>>>(psvbuf, wp1, b1, h1buf);
    conv3d_wmma_kernel<8, 0><<<blocks, 256, 0, stream>>>(h1buf,  wp2, b2, h2buf);
    conv3d_wmma_kernel<3, 1><<<blocks, 256, 0, stream>>>(h2buf,  wp3, b3, predbuf);
  }

  // 4) combine -> rgba (overwrites h2; h2 dead)
  combine_kernel<<<(VOX + 255)/256, 256, 0, stream>>>(predbuf, psvbuf, rgbabuf);

  // 5) forward splat (accumulator overwrites h1; h1 dead)
  {
    int n = 4*VOX;
    zero_kernel<<<(n + 255)/256, 256, 0, stream>>>(warpbuf, n);
    splat_kernel<<<(VOX + 255)/256, 256, 0, stream>>>(rgbabuf, Hfwd, warpbuf);
  }

  // 6) over-composite -> d_out
  composite_kernel<<<(HW + 255)/256, 256, 0, stream>>>(warpbuf, dispTab, (float*)d_out);
}
